// MultiHeadAttention_44504451121383
// MI455X (gfx1250) — matrix operand order repaired and checked by reading, not compiled, not run on hardware
//
#include <hip/hip_runtime.h>
#include <stdint.h>

// ---------------- types ----------------
typedef __attribute__((ext_vector_type(16))) __bf16 bf16x16;
typedef __attribute__((ext_vector_type(8)))  float  f32x8;
typedef __attribute__((ext_vector_type(4)))  unsigned int u32x4;
typedef __attribute__((ext_vector_type(4)))  float  f32x4;

static constexpr int SEQ = 2048;    // S
static constexpr int DMODEL = 1024; // D = H*DK
static constexpr int NHEAD = 16;    // H
static constexpr int HDIM = 64;     // DK = DV

__device__ __forceinline__ unsigned short f2bf(float f) {
    unsigned int u = __builtin_bit_cast(unsigned int, f);
    u += 0x7FFFu + ((u >> 16) & 1u);            // round to nearest even
    return (unsigned short)(u >> 16);
}
__device__ __forceinline__ unsigned int pack2(float a, float b) {
#if __has_builtin(__builtin_amdgcn_cvt_pk_bf16_f32)
    auto r = __builtin_amdgcn_cvt_pk_bf16_f32(a, b);   // v_cvt_pk_bf16_f32
    return __builtin_bit_cast(unsigned int, r);
#else
    return (unsigned int)f2bf(a) | ((unsigned int)f2bf(b) << 16);
#endif
}
__device__ __forceinline__ bf16x16 make_frag(u32x4 lo, u32x4 hi) {
    union { u32x4 i[2]; bf16x16 v; } u;
    u.i[0] = lo; u.i[1] = hi; return u.v;
}
__device__ __forceinline__ f32x8 fzero8() {
    f32x8 z;
#pragma unroll
    for (int i = 0; i < 8; ++i) z[i] = 0.0f;
    return z;
}
#define WMMA_BF16(A, B, C) \
    __builtin_amdgcn_wmma_f32_16x16x32_bf16(false, (A), false, (B), (short)0, (C), false, false)

// ---------------- weight transpose + f32->bf16 (one-time pre-pass) ----------------
// W: [1024][1024] f32  ->  WT: [n][k] bf16
__global__ void __launch_bounds__(256) wt_transpose_kernel(
        const float* __restrict__ W, unsigned short* __restrict__ WT) {
    int idx = blockIdx.x * 256 + threadIdx.x;  // k fastest in output -> coalesced writes
    int k = idx & 1023;
    int n = idx >> 10;
    WT[(size_t)(n << 10) + k] = f2bf(W[(size_t)(k << 10) + n]);
}

// ---------------- GEMM: C[M,N] = A[M,K](f32) * WT[N,K](bf16)^T + bias ----------------
// MODE 0: store bf16, head-split layout out[((b*H+h)*S + s)*64 + d]   (QKV projections)
// MODE 1: store f32, plain [m][n]                                      (output projection)
template <int MODE>
__global__ void __launch_bounds__(256) gemm_bias_kernel(
        const float* __restrict__ A, const unsigned short* __restrict__ WT,
        const float* __restrict__ bias, void* __restrict__ outp,
        int M, int N, int K) {
    __shared__ __align__(16) unsigned short ldsA[2][128 * 48]; // stride 48 (96B), double buffer
    __shared__ __align__(16) unsigned short ldsB[2][128 * 48]; // B stored N-major

    const int tid  = threadIdx.x;
    const int lane = tid & 31;
    const int wave = tid >> 5;         // 8 waves
    const int n0   = blockIdx.x * 128;
    const int m0   = blockIdx.y * 128;
    const int wm   = wave >> 2;        // 0..1  (M)
    const int wn   = wave & 3;         // 0..3  (N)
    const int half = lane >> 4;
    const int l16  = lane & 15;

    f32x8 acc[4][2];
#pragma unroll
    for (int i = 0; i < 4; ++i)
#pragma unroll
        for (int j = 0; j < 2; ++j) acc[i][j] = fzero8();

    const int srow  = tid >> 1;        // 0..127
    const int shalf = tid & 1;         // which 16-element half of BK=32
    const float*          aptr = A  + (size_t)(m0 + srow) * K + shalf * 16;
    const unsigned short* bptr = WT + (size_t)(n0 + srow) * K + shalf * 16;
    const int aoff = srow * 48 + shalf * 16;
    const int klo  = half ? 8 : 0;     // A frag: K chunks {klo..+7, klo+16..+23}
    const int klo2 = half ? 16 : 0;    // unused; B frag also uses klo (same split as A)

    // preload first tile into registers
    f32x4 a0 = ((const f32x4*)aptr)[0], a1 = ((const f32x4*)aptr)[1];
    f32x4 a2 = ((const f32x4*)aptr)[2], a3 = ((const f32x4*)aptr)[3];
    u32x4 bb0 = ((const u32x4*)bptr)[0], bb1 = ((const u32x4*)bptr)[1];

    int buf = 0;
    for (int kb = 0; kb < K; kb += 32, buf ^= 1) {
        // commit current tile regs -> LDS[buf]
        u32x4 p0, p1;
        p0[0] = pack2(a0.x, a0.y); p0[1] = pack2(a0.z, a0.w);
        p0[2] = pack2(a1.x, a1.y); p0[3] = pack2(a1.z, a1.w);
        p1[0] = pack2(a2.x, a2.y); p1[1] = pack2(a2.z, a2.w);
        p1[2] = pack2(a3.x, a3.y); p1[3] = pack2(a3.z, a3.w);
        *(u32x4*)&ldsA[buf][aoff]     = p0;
        *(u32x4*)&ldsA[buf][aoff + 8] = p1;
        *(u32x4*)&ldsB[buf][aoff]     = bb0;
        *(u32x4*)&ldsB[buf][aoff + 8] = bb1;
        __syncthreads();  // single barrier per K-step (double-buffered)

        // issue next tile's global loads now -> overlap with the WMMAs below
        if (kb + 32 < K) {
            const float*          an = aptr + kb + 32;
            const unsigned short* bn = bptr + kb + 32;
            a0 = ((const f32x4*)an)[0]; a1 = ((const f32x4*)an)[1];
            a2 = ((const f32x4*)an)[2]; a3 = ((const f32x4*)an)[3];
            bb0 = ((const u32x4*)bn)[0]; bb1 = ((const u32x4*)bn)[1];
            __builtin_prefetch(an + 32, 0, 1);   // warm L2 one more K-step ahead
            __builtin_prefetch(bn + 32, 0, 1);
        }

        // fragments + 8 WMMAs on LDS[buf]
        bf16x16 af[4], bfr[2];
#pragma unroll
        for (int mt = 0; mt < 4; ++mt) {
            int r = wm * 64 + mt * 16 + l16;
            af[mt] = make_frag(*(const u32x4*)&ldsA[buf][r * 48 + klo],
                               *(const u32x4*)&ldsA[buf][r * 48 + klo + 16]);
        }
#pragma unroll
        for (int nt = 0; nt < 2; ++nt) {
            int r = wn * 32 + nt * 16 + l16;
            bfr[nt] = make_frag(*(const u32x4*)&ldsB[buf][r * 48 + klo],
                                *(const u32x4*)&ldsB[buf][r * 48 + klo + 16]);
        }
#pragma unroll
        for (int mt = 0; mt < 4; ++mt)
#pragma unroll
            for (int nt = 0; nt < 2; ++nt)
                acc[mt][nt] = WMMA_BF16(af[mt], bfr[nt], acc[mt][nt]);
    }

    // epilogue: bias + store (strength-reduced 32-bit offsets from one base)
#pragma unroll
    for (int nt = 0; nt < 2; ++nt) {
        const int n  = n0 + wn * 32 + nt * 16 + l16;
        const float bv = bias[n];
        if (MODE == 0) {
            const int b  = m0 >> 11;                       // whole block in one batch
            const int s0 = (m0 & (SEQ - 1)) + wm * 64 + half * 8;
            const int h  = n >> 6;
            const int d  = n & (HDIM - 1);
            unsigned short* ob = (unsigned short*)outp
                + (((size_t)(b * NHEAD + h) * SEQ + s0) << 6) + d;
#pragma unroll
            for (int mt = 0; mt < 4; ++mt)
#pragma unroll
                for (int v = 0; v < 8; ++v)
                    ob[(mt * 16 + v) << 6] = f2bf(acc[mt][nt][v] + bv);
        } else {
            float* ob = (float*)outp + (size_t)(m0 + wm * 64 + half * 8) * N + n;
#pragma unroll
            for (int mt = 0; mt < 4; ++mt)
#pragma unroll
                for (int v = 0; v < 8; ++v)
                    ob[(mt * 16 + v) * N] = acc[mt][nt][v] + bv;
        }
    }
}

// ---------------- flash attention: per (b,h), bf16 WMMA, online softmax ----------------
// Qh/Kh/Vh: bf16 [b*H][S][64].  Z: f32 [b][s][H*64].
__global__ void __launch_bounds__(256) attn_kernel(
        const unsigned short* __restrict__ Qh, const unsigned short* __restrict__ Kh,
        const unsigned short* __restrict__ Vh, const unsigned char* __restrict__ mask,
        float* __restrict__ Z) {
    __shared__ __align__(16) unsigned short kls[32 * 72];    // K chunk, key-major
    __shared__ __align__(16) unsigned short vls[64 * 48];    // V chunk, dv-major (transposed)
    __shared__ __align__(16) unsigned short pls[8][16 * 48]; // per-wave P tile (bf16)
    __shared__ __align__(16) unsigned char  mls[8][16 * 32]; // per-wave mask tile

    const int tid  = threadIdx.x;
    const int lane = tid & 31;
    const int wave = tid >> 5;
    const int bh   = blockIdx.y;       // b*H + h
    const int b    = bh >> 4;
    const int h    = bh & (NHEAD - 1);
    const int q0   = blockIdx.x * 128 + wave * 16;   // 16 q-rows per wave
    const int half = lane >> 4;
    const int l16  = lane & 15;

    // Q fragments (A operand), straight from global: row = l16
    bf16x16 qf[2];
    {
        const unsigned short* qr = Qh + (((size_t)bh * SEQ + q0 + l16) << 6);
        const int qklo = half ? 8 : 0;
        qf[0] = make_frag(*(const u32x4*)(qr + qklo),      *(const u32x4*)(qr + qklo + 16));
        qf[1] = make_frag(*(const u32x4*)(qr + 32 + qklo), *(const u32x4*)(qr + 32 + qklo + 16));
    }

    f32x8 o[4];
#pragma unroll
    for (int f = 0; f < 4; ++f) o[f] = fzero8();
    float mrun[8], lrun[8];
#pragma unroll
    for (int v = 0; v < 8; ++v) { mrun[v] = -3.0e38f; lrun[v] = 0.0f; }

    const int key_t = tid >> 3;        // 0..31 : key row this thread stages
    const int d8    = (tid & 7) * 8;   // 0..56 : 8-element d chunk
    const size_t bh_off = ((size_t)bh * SEQ) << 6;
    const unsigned short* Kb = Kh + bh_off + ((size_t)key_t << 6) + d8;
    const unsigned short* Vb = Vh + bh_off + ((size_t)key_t << 6) + d8;
    const unsigned char*  Mb = mask + (size_t)b * SEQ * SEQ
                             + (size_t)(q0 + (lane >> 1)) * SEQ + (lane & 1) * 16;

    for (int kb = 0; kb < SEQ; kb += 32) {
        // stage K chunk [32 keys][64 d]
        *(u32x4*)&kls[key_t * 72 + d8] = *(const u32x4*)(Kb + ((size_t)kb << 6));
        // stage V chunk transposed -> vls[dv][key]
        {
            union { u32x4 v4; unsigned short u[8]; } uu;
            uu.v4 = *(const u32x4*)(Vb + ((size_t)kb << 6));
#pragma unroll
            for (int j = 0; j < 8; ++j) vls[(d8 + j) * 48 + key_t] = uu.u[j];
        }
        // stage this wave's 16x32 mask tile
        *(u32x4*)&mls[wave][(lane >> 1) * 32 + (lane & 1) * 16] = *(const u32x4*)(Mb + kb);
        if (kb + 32 < SEQ) {   // warm L2 for the next chunk
            __builtin_prefetch(Kb + ((size_t)(kb + 32) << 6), 0, 1);
            __builtin_prefetch(Vb + ((size_t)(kb + 32) << 6), 0, 1);
        }
        __syncthreads();

        // scores: two 16x16 tiles, each = 2 WMMAs over d
        f32x8 s[2];
        const int dlo = half ? 8 : 0;
#pragma unroll
        for (int kt = 0; kt < 2; ++kt) {
            int r = kt * 16 + l16;  // key row (B operand column)
            bf16x16 b0 = make_frag(*(const u32x4*)&kls[r * 72 + dlo],
                                   *(const u32x4*)&kls[r * 72 + dlo + 16]);
            bf16x16 b1 = make_frag(*(const u32x4*)&kls[r * 72 + 32 + dlo],
                                   *(const u32x4*)&kls[r * 72 + 32 + dlo + 16]);
            f32x8 z = fzero8();
            z = WMMA_BF16(qf[0], b0, z);
            z = WMMA_BF16(qf[1], b1, z);
            s[kt] = z;
        }
        // scale + mask (reference: logits/sqrt(DK), then where(mask, ., -1e15))
#pragma unroll
        for (int kt = 0; kt < 2; ++kt)
#pragma unroll
            for (int v = 0; v < 8; ++v) {
                float sv = s[kt][v] * 0.125f;
                unsigned char mb = mls[wave][(v + half * 8) * 32 + kt * 16 + l16];
                s[kt][v] = mb ? sv : -1.0e15f;
            }
        // online softmax (row = (v, lane-half); reduce across 16 lanes)
        float corr[8];
#pragma unroll
        for (int v = 0; v < 8; ++v) {
            float mx = fmaxf(s[0][v], s[1][v]);
#pragma unroll
            for (int off = 1; off < 16; off <<= 1)
                mx = fmaxf(mx, __shfl_xor(mx, off, 16));
            float mnew = fmaxf(mrun[v], mx);
            corr[v] = __expf(mrun[v] - mnew);
            float p0 = __expf(s[0][v] - mnew);
            float p1 = __expf(s[1][v] - mnew);
            s[0][v] = p0; s[1][v] = p1;
            float ps = p0 + p1;
#pragma unroll
            for (int off = 1; off < 16; off <<= 1)
                ps += __shfl_xor(ps, off, 16);
            lrun[v] = lrun[v] * corr[v] + ps;
            mrun[v] = mnew;
        }
#pragma unroll
        for (int f = 0; f < 4; ++f)
#pragma unroll
            for (int v = 0; v < 8; ++v) o[f][v] *= corr[v];

        // P (C-layout) -> LDS bf16 -> reload as A fragment
#pragma unroll
        for (int kt = 0; kt < 2; ++kt)
#pragma unroll
            for (int v = 0; v < 8; ++v)
                pls[wave][(v + half * 8) * 48 + kt * 16 + l16] = f2bf(s[kt][v]);
        const int pklo = half ? 8 : 0;
        bf16x16 pf = make_frag(*(const u32x4*)&pls[wave][l16 * 48 + pklo],
                               *(const u32x4*)&pls[wave][l16 * 48 + pklo + 16]);
        // P(16x32) * V(32x64): one WMMA per 16-wide dv tile
        const int kk = half ? 8 : 0;
#pragma unroll
        for (int f = 0; f < 4; ++f) {
            int dv = f * 16 + l16;
            bf16x16 vb = make_frag(*(const u32x4*)&vls[dv * 48 + kk],
                                   *(const u32x4*)&vls[dv * 48 + kk + 16]);
            o[f] = WMMA_BF16(pf, vb, o[f]);
        }
        __syncthreads();
    }

    // normalize + store z as f32 [b][s][h*64+dv]
    float inv[8];
#pragma unroll
    for (int v = 0; v < 8; ++v) inv[v] = 1.0f / lrun[v];
    float* zb = Z + ((size_t)b * SEQ + q0 + half * 8) * DMODEL + h * 64 + l16;
#pragma unroll
    for (int f = 0; f < 4; ++f)
#pragma unroll
        for (int v = 0; v < 8; ++v)
            zb[v * DMODEL + f * 16] = o[f][v] * inv[v];
}

// ---------------- host-side orchestration ----------------
extern "C" void kernel_launch(void* const* d_in, const int* in_sizes, int n_in,
                              void* d_out, int out_size, void* d_ws, size_t ws_size,
                              hipStream_t stream) {
    (void)in_sizes; (void)n_in; (void)out_size; (void)ws_size;
    const float* x_k = (const float*)d_in[0];
    const float* x_q = (const float*)d_in[1];
    const float* x_v = (const float*)d_in[2];
    const unsigned char* mask = (const unsigned char*)d_in[3];
    const float* Wk = (const float*)d_in[4];
    const float* bk = (const float*)d_in[5];
    const float* Wq = (const float*)d_in[6];
    const float* bq = (const float*)d_in[7];
    const float* Wv = (const float*)d_in[8];
    const float* bv = (const float*)d_in[9];
    const float* Wp = (const float*)d_in[10];
    const float* bp = (const float*)d_in[11];

    constexpr size_t MB = 1024ull * 1024ull;
    char* ws = (char*)d_ws;
    unsigned short* WkT = (unsigned short*)(ws + 0 * MB);
    unsigned short* WqT = (unsigned short*)(ws + 2 * MB);
    unsigned short* WvT = (unsigned short*)(ws + 4 * MB);
    unsigned short* WpT = (unsigned short*)(ws + 6 * MB);
    unsigned short* Qh  = (unsigned short*)(ws + 8 * MB);
    unsigned short* Kh  = (unsigned short*)(ws + 16 * MB);
    unsigned short* Vh  = (unsigned short*)(ws + 24 * MB);
    float*          Zws = (float*)(ws + 32 * MB);   // 16 MB -> 48 MB total

    const int M  = 2 * SEQ;   // B*S = 4096
    const int NK = DMODEL;    // 1024

    // one-time weight transpose/convert
    int tgrid = (NK * NK) / 256;
    wt_transpose_kernel<<<tgrid, 256, 0, stream>>>(Wq, WqT);
    wt_transpose_kernel<<<tgrid, 256, 0, stream>>>(Wk, WkT);
    wt_transpose_kernel<<<tgrid, 256, 0, stream>>>(Wv, WvT);
    wt_transpose_kernel<<<tgrid, 256, 0, stream>>>(Wp, WpT);

    // QKV projections (bf16 head-split outputs)
    dim3 ggrid(NK / 128, M / 128);
    gemm_bias_kernel<0><<<ggrid, 256, 0, stream>>>(x_q, WqT, bq, Qh, M, NK, NK);
    gemm_bias_kernel<0><<<ggrid, 256, 0, stream>>>(x_k, WkT, bk, Kh, M, NK, NK);
    gemm_bias_kernel<0><<<ggrid, 256, 0, stream>>>(x_v, WvT, bv, Vh, M, NK, NK);

    // flash attention: grid = (q-tiles, b*h)
    attn_kernel<<<dim3(SEQ / 128, 2 * NHEAD), 256, 0, stream>>>(Qh, Kh, Vh, mask, Zws);

    // output projection (f32 result)
    gemm_bias_kernel<1><<<ggrid, 256, 0, stream>>>(Zws, WpT, bp, d_out, M, NK, NK);
}
